// DCRNN_77240691851868
// MI455X (gfx1250) — matrix operand order repaired and checked by reading, not compiled, not run on hardware
//
#include <hip/hip_runtime.h>
#include <hip/hip_bf16.h>
#include <stdint.h>
#include <stddef.h>

// ---------------------------------------------------------------------------
// DCRNN on MI455X (gfx1250): bf16 WMMA GEMM pipeline.
//   N=1024 nodes, U=128 units, B=8, F=129, NM=5, SEQ=HOR=12.
// Dominant cost: S(1024x1024) @ X(1024x1032) x8 per cell, + XK(8192x645)@W.
// Working set fits in 192MB L2 -> compute-bound -> v_wmma_f32_16x16x32_bf16.
// GEMM: 128x128 block tile, 8 wave32 waves, double-buffered LDS (1 barrier
// per 32-deep K slice), global loads for slice k+1 issued before the WMMAs
// of slice k so matrix math hides memory latency.
// ---------------------------------------------------------------------------

typedef __bf16        bf16x16 __attribute__((ext_vector_type(16)));
typedef float         f32x8   __attribute__((ext_vector_type(8)));
typedef unsigned int  u32x4   __attribute__((ext_vector_type(4)));

#define NNODE   1024
#define UU      128
#define BBATCH  8
#define FFEAT   129          // din(1) + U(128)
#define LDX     1152         // padded F*B columns (9 * 128), zero-filled pad
#define LDKK    672          // padded F*NM (21 * 32), zero-filled pad
#define KVALID  645          // F * NM
#define SEQ_T   12
#define HOR_T   12

__device__ __forceinline__ unsigned short f2bf(float f) {
  unsigned u = __float_as_uint(f);
  u += 0x7FFFu + ((u >> 16) & 1u);          // round-to-nearest-even
  return (unsigned short)(u >> 16);
}
__device__ __forceinline__ float bf2f(unsigned short h) {
  return __uint_as_float(((unsigned)h) << 16);
}

union Frag { bf16x16 v; u32x4 u[2]; };

struct GemmArgs {
  const unsigned short* A;
  const unsigned short* B;
  unsigned short*       Cbf;    // EPI 0/1
  const unsigned short* X0;     // EPI 1
  const float*          bias;   // EPI 2/3
  const float*          hold;   // EPI 2/3
  float*                ubuf;   // EPI 2 write, EPI 3 read
  unsigned short*       x0c;    // EPI 2 write (candidate X0, (N, LDX) layout)
  float*                hnew;   // EPI 3 write
  int lda, ldb, ldc, Ktiles;
};

// Epilogues:
//   EPI 0: store bf16                         (x1 = S @ x0)
//   EPI 1: store bf16(2*acc - x0)             (x2 = 2*S@x1 - x0)
//   EPI 2: gate: s=sigmoid(acc+bias); col<U -> scatter r*h into X0c layout,
//          col>=U -> u buffer (fp32)
//   EPI 3: cand: c=tanh(acc+bias); hnew = u*hold + (1-u)*c
template <int EPI>
__global__ __launch_bounds__(256, 2)
void gemm_bf16_wmma(GemmArgs g) {
  __shared__ unsigned short As[2][128 * 40];   // A tile, row major, pad stride 40
  __shared__ unsigned short Bs[2][128 * 40];   // B tile TRANSPOSED: [col][k]

  const int tid   = threadIdx.x;
  const int lane  = tid & 31;
  const int wave  = tid >> 5;
  const int wrow  = wave & 3;               // 4 wave rows  (M)
  const int wcol  = wave >> 2;              // 2 wave cols  (N)
  const int mBase = blockIdx.y << 7;
  const int nBase = blockIdx.x << 7;
  const int l15   = lane & 15;
  const int hsel  = lane >> 4;

  // ---- per-thread staging coordinates (constant across K) ----
  const int ar0 = tid >> 2;                 // A rows ar0 and ar0+64
  const int kc0 = (tid & 3) << 3;
  const int kr0 = tid >> 4;                 // B rows kr0 and kr0+16
  const int nc0 = (tid & 15) << 3;
  const int sA0 = ar0 * 40 + kc0;
  const int sA1 = sA0 + 64 * 40;

  const unsigned short* pA0 = g.A + (size_t)(mBase + ar0) * g.lda + kc0;
  const unsigned short* pA1 = pA0 + (size_t)64 * g.lda;
  const unsigned short* pB0 = g.B + (size_t)kr0 * g.ldb + nBase + nc0;
  const unsigned short* pB1 = pB0 + (size_t)16 * g.ldb;
  const size_t bStep = (size_t)32 * g.ldb;

  // ---- fragment LDS offsets (constant across K) ----
  const int aOff = ((wrow << 5) + l15) * 40 + (hsel << 3);   // + mi*640
  const int bOff = ((wcol << 6) + l15) * 40 + (hsel << 3);   // + ni*640

  f32x8 acc[2][4];
#pragma unroll
  for (int i = 0; i < 2; ++i)
#pragma unroll
    for (int j = 0; j < 4; ++j)
#pragma unroll
      for (int e = 0; e < 8; ++e) acc[i][j][e] = 0.0f;

  // ---- prologue: tile 0 -> registers -> LDS buffer 0 ----
  u32x4 ra0 = *(const u32x4*)pA0;
  u32x4 ra1 = *(const u32x4*)pA1;
  u32x4 rb0 = *(const u32x4*)pB0;
  u32x4 rb1 = *(const u32x4*)pB1;
  pA0 += 32; pA1 += 32; pB0 += bStep; pB1 += bStep;
  {
    *(u32x4*)(&As[0][sA0]) = ra0;
    *(u32x4*)(&As[0][sA1]) = ra1;
    const unsigned short* s0 = (const unsigned short*)&rb0;
    const unsigned short* s1 = (const unsigned short*)&rb1;
#pragma unroll
    for (int j = 0; j < 8; ++j) {
      Bs[0][(nc0 + j) * 40 + kr0]      = s0[j];
      Bs[0][(nc0 + j) * 40 + kr0 + 16] = s1[j];
    }
  }

  for (int kt = 0; kt < g.Ktiles; ++kt) {
    const int  cur  = kt & 1;
    const bool more = (kt + 1) < g.Ktiles;
    if (more) {   // issue ALL global loads for slice kt+1 before the barrier
      ra0 = *(const u32x4*)pA0;
      ra1 = *(const u32x4*)pA1;
      rb0 = *(const u32x4*)pB0;
      rb1 = *(const u32x4*)pB1;
      pA0 += 32; pA1 += 32; pB0 += bStep; pB1 += bStep;
      __builtin_prefetch(pA0, 0, 1);     // global_prefetch_b8, slice kt+2
      __builtin_prefetch(pB0, 0, 1);
    }
    __syncthreads();                     // LDS buf[cur] now fully written

    const unsigned short* asBuf = &As[cur][0];
    const unsigned short* bsBuf = &Bs[cur][0];
    Frag af[2], bfr[4];
#pragma unroll
    for (int mi = 0; mi < 2; ++mi) {
      af[mi].u[0] = *(const u32x4*)(asBuf + aOff + mi * 640);
      af[mi].u[1] = *(const u32x4*)(asBuf + aOff + mi * 640 + 16);
    }
#pragma unroll
    for (int ni = 0; ni < 4; ++ni) {
      bfr[ni].u[0] = *(const u32x4*)(bsBuf + bOff + ni * 640);
      bfr[ni].u[1] = *(const u32x4*)(bsBuf + bOff + ni * 640 + 16);
    }
#pragma unroll
    for (int mi = 0; mi < 2; ++mi)
#pragma unroll
      for (int ni = 0; ni < 4; ++ni)
        acc[mi][ni] = __builtin_amdgcn_wmma_f32_16x16x32_bf16(
            false, af[mi].v, false, bfr[ni].v, (short)0, acc[mi][ni], false, false);

    if (more) {   // spill next slice into the other buffer (no barrier needed)
      unsigned short* asN = &As[cur ^ 1][0];
      unsigned short* bsN = &Bs[cur ^ 1][0];
      *(u32x4*)(asN + sA0) = ra0;
      *(u32x4*)(asN + sA1) = ra1;
      const unsigned short* s0 = (const unsigned short*)&rb0;
      const unsigned short* s1 = (const unsigned short*)&rb1;
#pragma unroll
      for (int j = 0; j < 8; ++j) {
        bsN[(nc0 + j) * 40 + kr0]      = s0[j];
        bsN[(nc0 + j) * 40 + kr0 + 16] = s1[j];
      }
    }
  }

  // ---- epilogue: C/D 16x16 f32 layout: lane n = l15, row m = 8*hsel + v ----
#pragma unroll
  for (int mi = 0; mi < 2; ++mi) {
#pragma unroll
    for (int ni = 0; ni < 4; ++ni) {
      const int tM = mBase + (wrow << 5) + (mi << 4);
      const int tN = nBase + (wcol << 6) + (ni << 4);
      const int n  = tN + l15;
#pragma unroll
      for (int v = 0; v < 8; ++v) {
        const int m = tM + (hsel << 3) + v;
        const float val = acc[mi][ni][v];
        if constexpr (EPI == 0) {
          g.Cbf[(size_t)m * g.ldc + n] = f2bf(val);
        } else if constexpr (EPI == 1) {
          const float x0 = bf2f(g.X0[(size_t)m * g.ldc + n]);
          g.Cbf[(size_t)m * g.ldc + n] = f2bf(2.0f * val - x0);
        } else if constexpr (EPI == 2) {
          const float s = 1.0f / (1.0f + __expf(-(val + g.bias[n])));
          const int b = m >> 10, nn = m & 1023;      // row = b*N + n_node
          if (n < UU) {
            const float rh = s * g.hold[(size_t)m * UU + n];
            g.x0c[(size_t)nn * LDX + (size_t)(1 + n) * BBATCH + b] = f2bf(rh);
          } else {
            g.ubuf[(size_t)m * UU + (n - UU)] = s;
          }
        } else {  // EPI == 3
          const float e = __expf(2.0f * (val + g.bias[n]));
          const float c = (e - 1.0f) / (e + 1.0f);   // tanh
          const size_t idx = (size_t)m * UU + n;
          const float u = g.ubuf[idx];
          g.hnew[idx] = u * g.hold[idx] + (1.0f - u) * c;
        }
      }
    }
  }
}

// ---------------------------------------------------------------------------
// Helper kernels
// ---------------------------------------------------------------------------
__global__ void dcrnn_degree_kernel(const float* __restrict__ adj,
                                    float* __restrict__ dinv,
                                    float* __restrict__ dcinv) {
  __shared__ float red[256];
  const int i = blockIdx.x;               // 0..1023 rows, 1024..2047 cols
  float s = 0.0f;
  if (i < NNODE) {
    for (int j = threadIdx.x; j < NNODE; j += 256) s += adj[(size_t)i * NNODE + j];
  } else {
    const int c = i - NNODE;
    for (int r = threadIdx.x; r < NNODE; r += 256) s += adj[(size_t)r * NNODE + c];
  }
  red[threadIdx.x] = s;
  __syncthreads();
  for (int off = 128; off > 0; off >>= 1) {
    if (threadIdx.x < off) red[threadIdx.x] += red[threadIdx.x + off];
    __syncthreads();
  }
  if (threadIdx.x == 0) {
    const float d = red[0];
    const float v = d > 0.0f ? 1.0f / d : 0.0f;
    if (i < NNODE) dinv[i] = v; else dcinv[i - NNODE] = v;
  }
}

// S1 = _rw(adj).T  : S1[i,j] = adj[j,i] * dinv[j]   (dinv = 1/rowsum)
// S2 = _rw(adj.T).T: S2[i,j] = adj[i,j] * dcinv[j]  (dcinv = 1/colsum)
__global__ void dcrnn_supports_kernel(const float* __restrict__ adj,
                                      const float* __restrict__ dinv,
                                      const float* __restrict__ dcinv,
                                      unsigned short* __restrict__ S1,
                                      unsigned short* __restrict__ S2) {
  const int idx = blockIdx.x * 256 + threadIdx.x;      // 1M
  const int i = idx >> 10, j = idx & 1023;
  S1[idx] = f2bf(adj[(size_t)j * NNODE + i] * dinv[j]);
  S2[idx] = f2bf(adj[(size_t)i * NNODE + j] * dcinv[j]);
}

__global__ void dcrnn_zero_kernel(float* __restrict__ p, int n) {
  const int i = blockIdx.x * 256 + threadIdx.x;
  if (i < n) p[i] = 0.0f;
}

// X0[n, f*B+b]: f==0 -> x[b,n]; f in 1..128 -> h[b, n*U + f-1]; pad -> 0.
// Writes both X0 and X0c (gate epilogue later overwrites X0c's h-part with r*h).
__global__ void dcrnn_build_x0_kernel(const float* __restrict__ xsrc, int xstride, int xoff,
                                      const float* __restrict__ h,
                                      unsigned short* __restrict__ X0,
                                      unsigned short* __restrict__ X0c) {
  const int idx = blockIdx.x * 256 + threadIdx.x;      // N * LDX
  const int n = idx / LDX, col = idx - n * LDX;
  unsigned short v = 0;
  if (col < FFEAT * BBATCH) {
    const int f = col >> 3, b = col & 7;
    float x;
    if (f == 0)
      x = xsrc ? xsrc[(size_t)((b << 10) + n) * xstride + xoff] : 0.0f;
    else
      x = h[(size_t)((b << 10) + n) * UU + (f - 1)];
    v = f2bf(x);
  }
  X0[idx] = v;
  X0c[idx] = v;
}

// XK[b*N+n, f*NM+m] = xs[m][n, f*B+b]; pad cols [645,672) -> 0
__global__ void dcrnn_build_xk_kernel(const unsigned short* __restrict__ x0,
                                      const unsigned short* __restrict__ x1a,
                                      const unsigned short* __restrict__ x2a,
                                      const unsigned short* __restrict__ x1b,
                                      const unsigned short* __restrict__ x2b,
                                      unsigned short* __restrict__ XK) {
  const int idx = blockIdx.x * 256 + threadIdx.x;      // 8192 * LDKK
  const int row = idx / LDKK, k = idx - row * LDKK;
  unsigned short v = 0;
  if (k < KVALID) {
    const int f = k / 5, m = k - f * 5;
    const int n = row & 1023, b = row >> 10;
    const unsigned short* p =
        (m == 0) ? x0 : (m == 1) ? x1a : (m == 2) ? x2a : (m == 3) ? x1b : x2b;
    v = p[(size_t)n * LDX + (size_t)f * BBATCH + b];
  }
  XK[idx] = v;
}

__global__ void dcrnn_convw_kernel(const float* __restrict__ W,
                                   unsigned short* __restrict__ Wbf, int outw) {
  const int idx = blockIdx.x * 256 + threadIdx.x;      // LDKK * outw
  const int r = idx / outw;
  Wbf[idx] = (r < KVALID) ? f2bf(W[idx]) : (unsigned short)0;
}

__global__ void dcrnn_proj_kernel(const float* __restrict__ h,
                                  const float* __restrict__ pw,
                                  const float* __restrict__ pb,
                                  float* __restrict__ out, int t) {
  const int row = blockIdx.x * 256 + threadIdx.x;      // 8192 = B*N
  if (row >= BBATCH * NNODE) return;
  float s = pb[0];
#pragma unroll 8
  for (int j = 0; j < UU; ++j) s += h[(size_t)row * UU + j] * pw[j];
  out[(size_t)row * HOR_T + t] = s;
}

// ---------------------------------------------------------------------------
// Host orchestration
// ---------------------------------------------------------------------------
extern "C" void kernel_launch(void* const* d_in, const int* in_sizes, int n_in,
                              void* d_out, int out_size, void* d_ws, size_t ws_size,
                              hipStream_t stream) {
  (void)in_sizes; (void)n_in; (void)out_size; (void)ws_size;
  const float* inputs = (const float*)d_in[0];
  /* labels d_in[1] unused */
  const float* adj    = (const float*)d_in[2];
  const float* enc_Wg = (const float*)d_in[3];
  const float* enc_bg = (const float*)d_in[4];
  const float* enc_Wc = (const float*)d_in[5];
  const float* enc_bc = (const float*)d_in[6];
  const float* dec_Wg = (const float*)d_in[7];
  const float* dec_bg = (const float*)d_in[8];
  const float* dec_Wc = (const float*)d_in[9];
  const float* dec_bc = (const float*)d_in[10];
  const float* proj_W = (const float*)d_in[11];
  const float* proj_b = (const float*)d_in[12];
  float* out = (float*)d_out;

  uintptr_t base = (uintptr_t)d_ws;
  auto carve = [&](size_t bytes) -> void* {
    uintptr_t p = (base + 255) & ~(uintptr_t)255;
    base = p + bytes;
    return (void*)p;
  };
  unsigned short* S1   = (unsigned short*)carve((size_t)NNODE * NNODE * 2);
  unsigned short* S2   = (unsigned short*)carve((size_t)NNODE * NNODE * 2);
  float* dinv          = (float*)carve(NNODE * 4);
  float* dcinv         = (float*)carve(NNODE * 4);
  unsigned short* X0   = (unsigned short*)carve((size_t)NNODE * LDX * 2);
  unsigned short* X0c  = (unsigned short*)carve((size_t)NNODE * LDX * 2);
  unsigned short* X1a  = (unsigned short*)carve((size_t)NNODE * LDX * 2);
  unsigned short* X2a  = (unsigned short*)carve((size_t)NNODE * LDX * 2);
  unsigned short* X1b  = (unsigned short*)carve((size_t)NNODE * LDX * 2);
  unsigned short* X2b  = (unsigned short*)carve((size_t)NNODE * LDX * 2);
  unsigned short* XK   = (unsigned short*)carve((size_t)BBATCH * NNODE * LDKK * 2);
  unsigned short* WgBf = (unsigned short*)carve((size_t)LDKK * 2 * UU * 2);
  unsigned short* WcBf = (unsigned short*)carve((size_t)LDKK * UU * 2);
  float* ubuf          = (float*)carve((size_t)BBATCH * NNODE * UU * 4);
  float* h_a           = (float*)carve((size_t)BBATCH * NNODE * UU * 4);
  float* h_b           = (float*)carve((size_t)BBATCH * NNODE * UU * 4);

  // --- supports (bf16) ---
  dcrnn_degree_kernel<<<2048, 256, 0, stream>>>(adj, dinv, dcinv);
  dcrnn_supports_kernel<<<4096, 256, 0, stream>>>(adj, dinv, dcinv, S1, S2);
  dcrnn_zero_kernel<<<4096, 256, 0, stream>>>(h_a, BBATCH * NNODE * UU);

  auto launch_cheb = [&](const unsigned short* S, const unsigned short* X,
                         unsigned short* Y, const unsigned short* X0sub) {
    GemmArgs a{};
    a.A = S;  a.lda = NNODE;
    a.B = X;  a.ldb = LDX;
    a.Cbf = Y; a.ldc = LDX; a.X0 = X0sub;
    a.Ktiles = NNODE / 32;
    if (X0sub) gemm_bf16_wmma<1><<<dim3(LDX / 128, NNODE / 128), 256, 0, stream>>>(a);
    else       gemm_bf16_wmma<0><<<dim3(LDX / 128, NNODE / 128), 256, 0, stream>>>(a);
  };

  auto run_cell = [&](const float* xsrc, int xstride, int xoff,
                      const unsigned short* Wg, const float* bg,
                      const unsigned short* Wc, const float* bc,
                      const float* hold, float* hnew) {
    dcrnn_build_x0_kernel<<<(NNODE * LDX) / 256, 256, 0, stream>>>(
        xsrc, xstride, xoff, hold, X0, X0c);
    // gate gconv: diffusion
    launch_cheb(S1, X0,  X1a, nullptr);
    launch_cheb(S1, X1a, X2a, X0);
    launch_cheb(S2, X0,  X1b, nullptr);
    launch_cheb(S2, X1b, X2b, X0);
    dcrnn_build_xk_kernel<<<(BBATCH * NNODE * LDKK) / 256, 256, 0, stream>>>(
        X0, X1a, X2a, X1b, X2b, XK);
    {   // gate GEMM + sigmoid; emits u and scatters r*h into X0c
      GemmArgs a{};
      a.A = XK; a.lda = LDKK;
      a.B = Wg; a.ldb = 2 * UU;
      a.Ktiles = LDKK / 32;
      a.bias = bg; a.hold = hold; a.ubuf = ubuf; a.x0c = X0c;
      gemm_bf16_wmma<2><<<dim3((2 * UU) / 128, (BBATCH * NNODE) / 128), 256, 0, stream>>>(a);
    }
    // candidate gconv: diffusion on [x, r*h]
    launch_cheb(S1, X0c, X1a, nullptr);
    launch_cheb(S1, X1a, X2a, X0c);
    launch_cheb(S2, X0c, X1b, nullptr);
    launch_cheb(S2, X1b, X2b, X0c);
    dcrnn_build_xk_kernel<<<(BBATCH * NNODE * LDKK) / 256, 256, 0, stream>>>(
        X0c, X1a, X2a, X1b, X2b, XK);
    {   // candidate GEMM + tanh + h update
      GemmArgs a{};
      a.A = XK; a.lda = LDKK;
      a.B = Wc; a.ldb = UU;
      a.Ktiles = LDKK / 32;
      a.bias = bc; a.hold = hold; a.ubuf = ubuf; a.hnew = hnew;
      gemm_bf16_wmma<3><<<dim3(UU / 128, (BBATCH * NNODE) / 128), 256, 0, stream>>>(a);
    }
  };

  float* hold = h_a;
  float* hnew = h_b;

  // --- encoder ---
  dcrnn_convw_kernel<<<(LDKK * 2 * UU) / 256, 256, 0, stream>>>(enc_Wg, WgBf, 2 * UU);
  dcrnn_convw_kernel<<<(LDKK * UU) / 256, 256, 0, stream>>>(enc_Wc, WcBf, UU);
  for (int t = 0; t < SEQ_T; ++t) {
    run_cell(inputs, SEQ_T, t, WgBf, enc_bg, WcBf, enc_bc, hold, hnew);
    float* tmp = hold; hold = hnew; hnew = tmp;
  }

  // --- decoder ---
  dcrnn_convw_kernel<<<(LDKK * 2 * UU) / 256, 256, 0, stream>>>(dec_Wg, WgBf, 2 * UU);
  dcrnn_convw_kernel<<<(LDKK * UU) / 256, 256, 0, stream>>>(dec_Wc, WcBf, UU);
  for (int t = 0; t < HOR_T; ++t) {
    const float* xsrc = (t == 0) ? nullptr : out;   // go-symbol zeros, then prev output
    run_cell(xsrc, HOR_T, t - 1, WgBf, dec_bg, WcBf, dec_bc, hold, hnew);
    dcrnn_proj_kernel<<<(BBATCH * NNODE) / 256, 256, 0, stream>>>(hnew, proj_W, proj_b, out, t);
    float* tmp = hold; hold = hnew; hnew = tmp;
  }
}
